// Context_Erased_Attention_Advanced_78941498900962
// MI455X (gfx1250) — matrix operand order repaired and checked by reading, not compiled, not run on hardware
//
#include <hip/hip_runtime.h>
#include <hip/hip_bf16.h>
#include <math.h>

// ---------------------------------------------------------------------------
// Problem constants (from reference)
// ---------------------------------------------------------------------------
#define BB      8192
#define KK      6
#define VDIM    2048
#define QDIM    1024
#define FDIM    (VDIM + QDIM)     // 3072 contraction dim
#define NHID    1024
#define SEIN    128
#define MROWS   (BB * KK)         // 49152

// GEMM tiling
#define MT      128               // rows per block (8 waves x 16 rows)
#define NT      128               // cols per block pass
#define FT      32                // K-chunk for wmma 16x16x32 bf16
#define NBLKS   (NHID / NT)       // 8
#define MTILES  (MROWS / MT)      // 384
#define FCHUNKS (FDIM / FT)       // 96
#define LSTR    40                // LDS row stride in ushort (80 B, 16B aligned)

typedef __attribute__((ext_vector_type(16))) __bf16 v16bf;
typedef __attribute__((ext_vector_type(8)))  float  v8f;

union Frag16 { v16bf v; unsigned p[8]; uint4 q[2]; };
union CFrag  { v8f v; float f[8]; };

__device__ __forceinline__ unsigned short f2bf(float x) {
    union { float f; unsigned u; } c; c.f = x;
    unsigned u = c.u;
    unsigned r = u + 0x7FFFu + ((u >> 16) & 1u);   // round-to-nearest-even
    return (unsigned short)(r >> 16);
}
__device__ __forceinline__ unsigned pack2(float a, float b) {
    return (unsigned)f2bf(a) | ((unsigned)f2bf(b) << 16);
}

// ---------------------------------------------------------------------------
// Kernel 0: zero the logits accumulator in workspace
// ---------------------------------------------------------------------------
__global__ void init_ws(float* logits) {
    int i = blockIdx.x * blockDim.x + threadIdx.x;
    if (i < MROWS) logits[i] = 0.0f;
}

// ---------------------------------------------------------------------------
// Kernel 1: fused GEMM  logits[m] += sum_h relu(vq[m,:].W1[h,:] + b1[h])*Wl[h]
//   grid = (NBLKS fast, MTILES), block = 256 (8 wave32)
//   double-buffered LDS; float4 global staging; b128 LDS fragment loads
// ---------------------------------------------------------------------------
__device__ __forceinline__ void load_chunk(const float* __restrict__ v,
                                           const float* __restrict__ q,
                                           const float* __restrict__ W1,
                                           int mBase, int nBase, int fBase,
                                           int sRow, int sCol,
                                           float4* ra, float4* rb) {
#pragma unroll
    for (int p = 0; p < 4; ++p) {
        const int row = sRow + p * 32;
        const int m   = mBase + row;
        const int bIx = m / KK;
        const int kIx = m - bIx * KK;
        const float* src = (fBase < VDIM)
            ? v + ((size_t)bIx * KK + kIx) * VDIM + fBase
            : q + (size_t)bIx * QDIM + (fBase - VDIM);
        ra[p] = *(const float4*)(src + sCol);
        rb[p] = *(const float4*)(W1 + (size_t)(nBase + row) * FDIM + fBase + sCol);
    }
}

__global__ __launch_bounds__(256)
void gemm_logits(const float* __restrict__ v,
                 const float* __restrict__ q,
                 const float* __restrict__ W1,
                 const float* __restrict__ b1,
                 const float* __restrict__ Wl,
                 float* __restrict__ logits) {
    __shared__ __align__(16) unsigned short ldsA[2][MT * LSTR];
    __shared__ __align__(16) unsigned short ldsB[2][NT * LSTR];

    const int tid   = threadIdx.x;
    const int lane  = tid & 31;
    const int wave  = tid >> 5;                  // 0..7, owns rows wave*16..+15
    const int nBase = blockIdx.x * NT;
    const int mBase = blockIdx.y * MT;

    CFrag acc[NT / 16];
#pragma unroll
    for (int nf = 0; nf < NT / 16; ++nf)
#pragma unroll
        for (int i = 0; i < 8; ++i) acc[nf].f[i] = 0.0f;

    // staging thread mapping: 32 cols / 4 = 8 float4 per row, 32 rows/pass
    const int sRow = tid >> 3;                   // 0..31
    const int sCol = (tid & 7) * 4;              // 0,4,...,28

    // fragment addressing (per wave / lane)
    const int aRow = (wave << 4) + (lane & 15);
    const int kb   = (lane < 16) ? 0 : 8;        // A: K-half select (ushorts)
    const int kh   = (lane < 16) ? 0 : 8;        // B: K-half select (ushorts)
    const int col  = lane & 15;                  // B: column within 16-group

    float4 ra[4], rb[4];
    load_chunk(v, q, W1, mBase, nBase, 0, sRow, sCol, ra, rb);

    for (int fc = 0; fc < FCHUNKS; ++fc) {
        const int buf = fc & 1;

        // ---- commit staged registers to LDS (bf16, packed b64 stores)
#pragma unroll
        for (int p = 0; p < 4; ++p) {
            const int row = sRow + p * 32;
            unsigned long long da =
                (unsigned long long)pack2(ra[p].x, ra[p].y) |
                ((unsigned long long)pack2(ra[p].z, ra[p].w) << 32);
            *(unsigned long long*)&ldsA[buf][row * LSTR + sCol] = da;
            unsigned long long db =
                (unsigned long long)pack2(rb[p].x, rb[p].y) |
                ((unsigned long long)pack2(rb[p].z, rb[p].w) << 32);
            *(unsigned long long*)&ldsB[buf][row * LSTR + sCol] = db;
        }
        __syncthreads();

        // ---- prefetch next chunk into registers (overlaps with WMMA below)
        if (fc + 1 < FCHUNKS)
            load_chunk(v, q, W1, mBase, nBase, (fc + 1) * FT, sRow, sCol, ra, rb);

        // ---- A fragment (16x32 bf16, ISA layout): two b128 LDS reads
        Frag16 af;
        {
            const unsigned short* ap = &ldsA[buf][aRow * LSTR];
            af.q[0] = *(const uint4*)(ap + kb);        // K = kb..kb+7
            af.q[1] = *(const uint4*)(ap + 16 + kb);   // K = 16+kb..16+kb+7
        }
        // ---- 8 B fragments (32x16 bf16) + 8 WMMAs
#pragma unroll
        for (int nf = 0; nf < NT / 16; ++nf) {
            Frag16 bfr;
            const unsigned short* bp = &ldsB[buf][(nf * 16 + col) * LSTR + kh];
            bfr.q[0] = *(const uint4*)bp;              // K = kh..kh+7
            bfr.q[1] = *(const uint4*)(bp + 16);       // K = 16+kh..16+kh+7
            acc[nf].v = __builtin_amdgcn_wmma_f32_16x16x32_bf16(
                false, af.v, false, bfr.v, (short)0, acc[nf].v, false, false);
        }
        // no trailing barrier: next iteration writes the other LDS buffer
    }

    // ---- epilogue: +b1, ReLU, *Wl, reduce 128 cols -> per-row partial logit
    float rs[8];
#pragma unroll
    for (int i = 0; i < 8; ++i) rs[i] = 0.0f;
#pragma unroll
    for (int nf = 0; nf < NT / 16; ++nf) {
        const int h    = nBase + nf * 16 + (lane & 15);
        const float bv = b1[h];
        const float wv = Wl[h];
#pragma unroll
        for (int i = 0; i < 8; ++i) {
            float t = acc[nf].f[i] + bv;
            t = fmaxf(t, 0.0f);
            rs[i] += t * wv;
        }
    }
    // reduce across the 16 lanes of each C-matrix half (wave32)
#pragma unroll
    for (int off = 8; off >= 1; off >>= 1)
#pragma unroll
        for (int i = 0; i < 8; ++i)
            rs[i] += __shfl_xor(rs[i], off, 32);

    if ((lane & 15) == 0) {
        const int rowOff = (lane < 16) ? 0 : 8;     // VGPR i -> row i / i+8
        const int m0 = mBase + (wave << 4) + rowOff;
#pragma unroll
        for (int i = 0; i < 8; ++i)
            atomicAdd(&logits[m0 + i], rs[i]);
    }
}

// ---------------------------------------------------------------------------
// Kernel 2: per-batch softmax + context-erase + SE gate -> scalars s[b,k]
//   one thread per batch element b
// ---------------------------------------------------------------------------
__global__ __launch_bounds__(256)
void att_gate(const float* __restrict__ logits,
              const float* __restrict__ mask,
              const float* __restrict__ bl,
              const float* __restrict__ Wse1,
              const float* __restrict__ Wse2,
              float* __restrict__ single_att,
              float* __restrict__ s_out) {
    __shared__ float s1[SEIN], s2[SEIN];
    if (threadIdx.x < SEIN) {
        s1[threadIdx.x] = Wse1[threadIdx.x];
        s2[threadIdx.x] = Wse2[threadIdx.x];
    }
    __syncthreads();

    const int b = blockIdx.x * blockDim.x + threadIdx.x;
    if (b >= BB) return;
    const float blv = bl[0];

    float l[KK];
    float mx = -1e30f;
#pragma unroll
    for (int k = 0; k < KK; ++k) {
        l[k] = logits[b * KK + k] + blv;
        mx = fmaxf(mx, l[k]);
    }
    float se = 0.0f, ex[KK];
#pragma unroll
    for (int k = 0; k < KK; ++k) { ex[k] = expf(l[k] - mx); se += ex[k]; }
    const float inv = 1.0f / se;

    float wc[KK], tot = 0.0f;
#pragma unroll
    for (int k = 0; k < KK; ++k) {
        wc[k] = l[k] * mask[b * KK + k];
        tot  += wc[k];
    }
#pragma unroll
    for (int k = 0; k < KK; ++k) {
        single_att[b * KK + k] = ex[k] * inv;
        const float x = tot - wc[k];                 // neigh
        float t = 0.0f;
        for (int i = 0; i < SEIN; ++i)
            t += s2[i] * fmaxf(x * s1[i], 0.0f);
        const float y = 1.0f / (1.0f + expf(-t));    // sigmoid
        s_out[b * KK + k] = x * (1.0f - y) + l[k];   // gated + w
    }
}

// ---------------------------------------------------------------------------
// Kernel 3: out2[mk, :] = s[mk] * Wf + bf   (402 MB write, float4 stores)
// ---------------------------------------------------------------------------
__global__ __launch_bounds__(256)
void outer_write(const float* __restrict__ s_in,
                 const float* __restrict__ Wf,
                 const float* __restrict__ bf,
                 float* __restrict__ out2) {
    const int idx = blockIdx.x * blockDim.x + threadIdx.x;   // vec4 id
    const int mk  = idx >> 9;                                // / (2048/4)
    const int dv  = (idx & 511) << 2;
    const float s = s_in[mk];
    const float4 wf = *(const float4*)(Wf + dv);
    const float4 bv = *(const float4*)(bf + dv);
    float4 o;
    o.x = s * wf.x + bv.x;
    o.y = s * wf.y + bv.y;
    o.z = s * wf.z + bv.z;
    o.w = s * wf.w + bv.w;
    *(float4*)(out2 + (size_t)mk * VDIM + dv) = o;
}

// ---------------------------------------------------------------------------
extern "C" void kernel_launch(void* const* d_in, const int* in_sizes, int n_in,
                              void* d_out, int out_size, void* d_ws, size_t ws_size,
                              hipStream_t stream) {
    const float* v    = (const float*)d_in[0];
    const float* q    = (const float*)d_in[1];
    const float* mask = (const float*)d_in[2];
    const float* W1   = (const float*)d_in[3];
    const float* b1   = (const float*)d_in[4];
    const float* Wl   = (const float*)d_in[5];
    const float* bl   = (const float*)d_in[6];
    const float* Wse1 = (const float*)d_in[7];
    const float* Wse2 = (const float*)d_in[8];
    const float* Wf   = (const float*)d_in[9];
    const float* bf   = (const float*)d_in[10];

    float* single_att = (float*)d_out;               // [B*K]
    float* out2       = (float*)d_out + MROWS;       // [B*K, VDIM]

    float* ws_logits = (float*)d_ws;                 // [MROWS]
    float* ws_s      = ws_logits + MROWS;            // [MROWS]

    init_ws<<<(MROWS + 255) / 256, 256, 0, stream>>>(ws_logits);

    dim3 ggrid(NBLKS, MTILES);                       // n fastest -> A reuse in L2
    gemm_logits<<<ggrid, 256, 0, stream>>>(v, q, W1, b1, Wl, ws_logits);

    att_gate<<<(BB + 255) / 256, 256, 0, stream>>>(ws_logits, mask, bl,
                                                   Wse1, Wse2, single_att, ws_s);

    const int nvec4 = MROWS * (VDIM / 4);            // 25,165,824
    outer_write<<<nvec4 / 256, 256, 0, stream>>>(ws_s, Wf, bf, out2);
}
